// SoftMoE_25967372272342
// MI455X (gfx1250) — compile-verified, matrix operand order checked by reading, not run on hardware
//
#include <hip/hip_runtime.h>
#include <cstdint>
#include <cstddef>

typedef __bf16 bf16;
typedef __bf16 v16bf __attribute__((ext_vector_type(16)));
typedef __bf16 v4bf  __attribute__((ext_vector_type(4)));
typedef float  v8f   __attribute__((ext_vector_type(8)));

#define DMODEL 1024
#define SEQ    2048
#define BATCH  4
#define NSLOT  16
#define FFN    4096
#define TOK    (BATCH*SEQ)        // 8192

// ---------------------------------------------------------------- utilities
__device__ __forceinline__ float silu_f(float x) { return x / (1.f + __expf(-x)); }

// 32-bit LDS offset from a generic pointer to __shared__ (low 32 bits of the
// LDS aperture address are the LDS byte offset on gfx1250).
__device__ __forceinline__ unsigned lds_off32(const void* p)
{
    return (unsigned)(uintptr_t)p;
}

// Async DMA of one 32-byte (row,half) segment of each tile straight into LDS.
// Tracked by ASYNCcnt; IOFFSET is added to BOTH the global and LDS addresses,
// which matches our mirrored (row*LDT + half) layout.
__device__ __forceinline__ void async_tile_load(unsigned ldsA, unsigned ldsB,
                                                const bf16* ga, const bf16* gb)
{
    asm volatile(
        "global_load_async_to_lds_b128 %0, %2, off\n\t"
        "global_load_async_to_lds_b128 %0, %2, off offset:16\n\t"
        "global_load_async_to_lds_b128 %1, %3, off\n\t"
        "global_load_async_to_lds_b128 %1, %3, off offset:16"
        :: "v"(ldsA), "v"(ldsB), "v"(ga), "v"(gb)
        : "memory");
}

__device__ __forceinline__ void wait_asynccnt0()
{
    asm volatile("s_wait_asynccnt 0x0" ::: "memory");
}

// ------------------------------------------------------- fp32 -> bf16 (flat)
__global__ __launch_bounds__(256)
void k_cvt_bf16(const float* __restrict__ src, bf16* __restrict__ dst, int n4)
{
    int i = blockIdx.x * 256 + threadIdx.x;
    if (i >= n4) return;
    float4 v = ((const float4*)src)[i];
    v4bf o; o[0] = (bf16)v.x; o[1] = (bf16)v.y; o[2] = (bf16)v.z; o[3] = (bf16)v.w;
    *(v4bf*)(dst + (size_t)i * 4) = o;
}

// -------------------------------------- W (K x N) fp32 -> W^T (N x K) bf16
__global__ __launch_bounds__(256)
void k_transpose_bf16(const float* __restrict__ W, bf16* __restrict__ Wt, int K, int N)
{
    __shared__ float tile[32][33];
    int nb = blockIdx.x * 32, kb = blockIdx.y * 32;
    int tx = threadIdx.x & 31, ty = threadIdx.x >> 5;   // 8 rows per pass
    for (int i = ty; i < 32; i += 8)
        tile[i][tx] = W[(size_t)(kb + i) * N + nb + tx];
    __syncthreads();
    for (int i = ty; i < 32; i += 8)
        Wt[(size_t)(nb + i) * K + kb + tx] = (bf16)tile[tx][i];
}

// ------------------------------------------------------- bf16 WMMA GEMM
// C[M,N] (fp32) = A[M,K](bf16, row-major) * Bt[N,K]^T (bf16, N-major) + bias
// Block tile 128x128, BK=32, 8 waves (4 M-groups x 2 N-groups), wave = 32x64.
// Double-buffered LDS filled by GLOBAL_LOAD_ASYNC_TO_LDS_B128 (ASYNCcnt),
// one barrier per K-step; DMA of tile k+1 overlaps the 8 WMMAs of tile k.
#define BM 128
#define BN 128
#define BK 32
#define LDT 40   // LDS row stride (bf16) -> 80B rows, 16B aligned halves

__global__ __launch_bounds__(256)
void k_gemm_bf16(const bf16* __restrict__ A, const bf16* __restrict__ Bt,
                 const float* __restrict__ bias, float* __restrict__ C,
                 int M, int N, int K)
{
    __shared__ __align__(16) bf16 sA[2][BM * LDT];
    __shared__ __align__(16) bf16 sB[2][BN * LDT];

    const int tid  = threadIdx.x;
    const int lane = tid & 31;
    const int wave = tid >> 5;
    const int bm = blockIdx.y * BM;
    const int bn = blockIdx.x * BN;
    const int wm = (wave & 3) * 32;     // wave M offset inside tile
    const int wn = (wave >> 2) * 64;    // wave N offset inside tile
    const int lrow = lane & 15;
    const int kh   = (lane >> 4) * 16;  // K-half select: lane half h holds contiguous k = 16h..16h+15 (A and B alike)

    const int lr = tid >> 1;            // 0..127 : tile row loaded by this thread
    const int lh = (tid & 1) * 16;      // 0 / 16 : K-half loaded

    const bf16* gA = A  + (size_t)(bm + lr) * K + lh;
    const bf16* gB = Bt + (size_t)(bn + lr) * K + lh;
    const unsigned ldsA0 = lds_off32(&sA[0][lr * LDT + lh]);
    const unsigned ldsA1 = lds_off32(&sA[1][lr * LDT + lh]);
    const unsigned ldsB0 = lds_off32(&sB[0][lr * LDT + lh]);
    const unsigned ldsB1 = lds_off32(&sB[1][lr * LDT + lh]);

    v8f acc[2][4] = {};

    // prologue: DMA tile 0 into buffer 0
    async_tile_load(ldsA0, ldsB0, gA, gB);

    for (int k0 = 0; k0 < K; k0 += BK) {
        const int buf = (k0 / BK) & 1;
        // my DMA into `buf` is done; barrier => everyone's DMA into `buf` is
        // done AND everyone has finished reading `buf^1` (previous iteration).
        wait_asynccnt0();
        __syncthreads();
        if (k0 + BK < K) {
            if (buf == 0) async_tile_load(ldsA1, ldsB1, gA + k0 + BK, gB + k0 + BK);
            else          async_tile_load(ldsA0, ldsB0, gA + k0 + BK, gB + k0 + BK);
        }

        v16bf af[2], bfr[4];
        #pragma unroll
        for (int mi = 0; mi < 2; ++mi) {
            const bf16* p = &sA[buf][(wm + mi * 16 + lrow) * LDT + kh];
            ((int4*)&af[mi])[0] = *(const int4*)p;
            ((int4*)&af[mi])[1] = *(const int4*)(p + 8);
        }
        #pragma unroll
        for (int ni = 0; ni < 4; ++ni) {
            const bf16* p = &sB[buf][(wn + ni * 16 + lrow) * LDT + kh];
            ((int4*)&bfr[ni])[0] = *(const int4*)p;
            ((int4*)&bfr[ni])[1] = *(const int4*)(p + 8);
        }
        #pragma unroll
        for (int mi = 0; mi < 2; ++mi)
            #pragma unroll
            for (int ni = 0; ni < 4; ++ni)
                acc[mi][ni] = __builtin_amdgcn_wmma_f32_16x16x32_bf16(
                    false, af[mi], false, bfr[ni], (short)0, acc[mi][ni], false, false);
    }

    // C/D layout: VGPR r, lanes 0-15 -> M=r, lanes 16-31 -> M=r+8; N = lane%16
    const int rbase = (lane >> 4) * 8;
    const int col0  = lane & 15;
    #pragma unroll
    for (int mi = 0; mi < 2; ++mi)
        #pragma unroll
        for (int ni = 0; ni < 4; ++ni) {
            const int col = bn + wn + ni * 16 + col0;
            const float bv = bias[col];
            #pragma unroll
            for (int r = 0; r < 8; ++r) {
                const int row = bm + wm + mi * 16 + rbase + r;
                C[(size_t)row * N + col] = acc[mi][ni][r] + bv;
            }
        }
}

// ------------------------------------------- small fp32 GEMM (M<=64 rows)
// C[m,n] = A[m,:] . W[:,n] + bias[n] (+ res[m % res_mod, n])
__global__ __launch_bounds__(256)
void k_small_gemm(const float* __restrict__ A, const float* __restrict__ W,
                  const float* __restrict__ bias, const float* __restrict__ res,
                  int res_mod, float* __restrict__ C, int N, int K)
{
    const int n = blockIdx.x * 256 + threadIdx.x;
    const int m = blockIdx.y;
    float acc = bias[n];
    const float* a = A + (size_t)m * K;
    const float* w = W + n;
    #pragma unroll 4
    for (int k = 0; k < K; ++k) acc += a[k] * w[(size_t)k * N];
    if (res) acc += res[(size_t)(m % res_mod) * N + n];
    C[(size_t)m * N + n] = acc;
}

// --------------------------------------- dispatch attention (flash softmax)
// per block: one (b,h). 16 slot queries attend over 2048 tokens.
__global__ __launch_bounds__(256)
void k_dispatch_attn(const float* __restrict__ qh,   // 16 x D
                     const float* __restrict__ Kp,   // TOK x D
                     const float* __restrict__ Vp,   // TOK x D
                     float* __restrict__ attn_d)     // (B*16) x D
{
    __shared__ __align__(16) float q[16][64];
    __shared__ float sc[16][257];
    __shared__ float red[16][17];
    __shared__ float runM[16], runL[16], fS[16], cM[16];

    const int bh = blockIdx.x, b = bh >> 4, h = bh & 15;
    const int tid = threadIdx.x;

    for (int i = tid; i < 1024; i += 256)
        q[i >> 6][i & 63] = qh[(size_t)(i >> 6) * DMODEL + h * 64 + (i & 63)];
    if (tid < 16) { runM[tid] = -3.0e38f; runL[tid] = 0.f; }
    __syncthreads();

    const int d  = tid & 63;
    const int sg = (tid >> 6) * 4;           // this thread accumulates slots sg..sg+3
    float acc0 = 0.f, acc1 = 0.f, acc2 = 0.f, acc3 = 0.f;

    for (int kc = 0; kc < SEQ; kc += 256) {
        {   // raw scores for token kc+tid, all 16 slots
            const float4* krow = (const float4*)(Kp + (size_t)(b * SEQ + kc + tid) * DMODEL + h * 64);
            float a[16];
            #pragma unroll
            for (int s = 0; s < 16; ++s) a[s] = 0.f;
            for (int i = 0; i < 16; ++i) {
                const float4 kv = krow[i];
                #pragma unroll
                for (int s = 0; s < 16; ++s) {
                    const float4 qv = ((const float4*)q[s])[i];
                    a[s] += qv.x * kv.x + qv.y * kv.y + qv.z * kv.z + qv.w * kv.w;
                }
            }
            #pragma unroll
            for (int s = 0; s < 16; ++s) sc[s][tid] = a[s] * 0.125f;
        }
        __syncthreads();
        {   // chunk max (16 threads per slot)
            const int s = tid >> 4, j = tid & 15;
            float mx = -3.0e38f;
            for (int kk = j; kk < 256; kk += 16) mx = fmaxf(mx, sc[s][kk]);
            red[s][j] = mx;
        }
        __syncthreads();
        if (tid < 16) {
            float mx = -3.0e38f;
            for (int j = 0; j < 16; ++j) mx = fmaxf(mx, red[tid][j]);
            const float nM = fmaxf(runM[tid], mx);
            fS[tid] = __expf(runM[tid] - nM);
            cM[tid] = nM; runM[tid] = nM;
        }
        __syncthreads();
        {   // exponentiate + chunk sum
            const int s = tid >> 4, j = tid & 15;
            const float nM = cM[s];
            float sm = 0.f;
            for (int kk = j; kk < 256; kk += 16) {
                const float p = __expf(sc[s][kk] - nM);
                sc[s][kk] = p; sm += p;
            }
            red[s][j] = sm;
        }
        __syncthreads();
        if (tid < 16) {
            float sm = 0.f;
            for (int j = 0; j < 16; ++j) sm += red[tid][j];
            runL[tid] = runL[tid] * fS[tid] + sm;
        }
        __syncthreads();
        {   // rescale + accumulate V
            acc0 *= fS[sg + 0]; acc1 *= fS[sg + 1]; acc2 *= fS[sg + 2]; acc3 *= fS[sg + 3];
            const float* vcol = Vp + (size_t)(b * SEQ + kc) * DMODEL + h * 64 + d;
            for (int kk = 0; kk < 256; ++kk) {
                const float vv = vcol[(size_t)kk * DMODEL];
                acc0 += sc[sg + 0][kk] * vv;
                acc1 += sc[sg + 1][kk] * vv;
                acc2 += sc[sg + 2][kk] * vv;
                acc3 += sc[sg + 3][kk] * vv;
            }
        }
        __syncthreads();
    }
    attn_d[(size_t)(b * 16 + sg + 0) * DMODEL + h * 64 + d] = acc0 / runL[sg + 0];
    attn_d[(size_t)(b * 16 + sg + 1) * DMODEL + h * 64 + d] = acc1 / runL[sg + 1];
    attn_d[(size_t)(b * 16 + sg + 2) * DMODEL + h * 64 + d] = acc2 / runL[sg + 2];
    attn_d[(size_t)(b * 16 + sg + 3) * DMODEL + h * 64 + d] = acc3 / runL[sg + 3];
}

// ------------------------------------------------ per-slot LayerNorm (fp32)
__global__ __launch_bounds__(256)
void k_expert_ln(const float* __restrict__ o, const float* __restrict__ g,
                 const float* __restrict__ be, float* __restrict__ hn)
{
    __shared__ float red[256];
    __shared__ float s_mu, s_rs;
    const int row = blockIdx.x, e = row & 15, tid = threadIdx.x;
    const float* src = o + (size_t)row * DMODEL;
    float v[4]; float s = 0.f;
    #pragma unroll
    for (int i = 0; i < 4; ++i) { v[i] = src[tid + i * 256]; s += v[i]; }
    red[tid] = s; __syncthreads();
    for (int st = 128; st > 0; st >>= 1) { if (tid < st) red[tid] += red[tid + st]; __syncthreads(); }
    if (tid == 0) s_mu = red[0] * (1.f / 1024.f);
    __syncthreads();
    const float mu = s_mu;
    float vs = 0.f;
    #pragma unroll
    for (int i = 0; i < 4; ++i) { const float dd = v[i] - mu; vs += dd * dd; }
    red[tid] = vs; __syncthreads();
    for (int st = 128; st > 0; st >>= 1) { if (tid < st) red[tid] += red[tid + st]; __syncthreads(); }
    if (tid == 0) s_rs = rsqrtf(red[0] * (1.f / 1024.f) + 1e-5f);
    __syncthreads();
    const float rs = s_rs;
    #pragma unroll
    for (int i = 0; i < 4; ++i) {
        const int dd = tid + i * 256;
        hn[(size_t)row * DMODEL + dd] = (v[i] - mu) * rs * g[e * DMODEL + dd] + be[e * DMODEL + dd];
    }
}

// ------------------------ expert MLP stage 1: m = silu(h W1+b1)*silu(h Wg+bg)
// grid (F/256, E); streams 536 MB of W1/Wg exactly once (bandwidth bound).
__global__ __launch_bounds__(256)
void k_expert_mlp1(const float* __restrict__ hn, const float* __restrict__ W1,
                   const float* __restrict__ b1, const float* __restrict__ Wg,
                   const float* __restrict__ bg, float* __restrict__ mbuf)
{
    __shared__ float h[4][DMODEL];
    const int e = blockIdx.y, tid = threadIdx.x;
    const int f = blockIdx.x * 256 + tid;
    for (int i = tid; i < 4 * DMODEL; i += 256)
        h[i >> 10][i & 1023] = hn[(size_t)((i >> 10) * 16 + e) * DMODEL + (i & 1023)];
    __syncthreads();

    const float* w1 = W1 + (size_t)e * DMODEL * FFN + f;
    const float* wg = Wg + (size_t)e * DMODEL * FFN + f;
    float au0 = 0, au1 = 0, au2 = 0, au3 = 0, ag0 = 0, ag1 = 0, ag2 = 0, ag3 = 0;
    #pragma unroll 4
    for (int dd = 0; dd < DMODEL; ++dd) {
        const float a = w1[(size_t)dd * FFN];
        const float g = wg[(size_t)dd * FFN];
        const float h0 = h[0][dd], h1 = h[1][dd], h2 = h[2][dd], h3 = h[3][dd];
        au0 += h0 * a; au1 += h1 * a; au2 += h2 * a; au3 += h3 * a;
        ag0 += h0 * g; ag1 += h1 * g; ag2 += h2 * g; ag3 += h3 * g;
    }
    const float bu = b1[e * FFN + f], bv = bg[e * FFN + f];
    mbuf[(size_t)(0 * 16 + e) * FFN + f] = silu_f(au0 + bu) * silu_f(ag0 + bv);
    mbuf[(size_t)(1 * 16 + e) * FFN + f] = silu_f(au1 + bu) * silu_f(ag1 + bv);
    mbuf[(size_t)(2 * 16 + e) * FFN + f] = silu_f(au2 + bu) * silu_f(ag2 + bv);
    mbuf[(size_t)(3 * 16 + e) * FFN + f] = silu_f(au3 + bu) * silu_f(ag3 + bv);
}

// ----------------- expert MLP stage 2 partials: part[fc,row,d] = m.W2 chunk
// grid (D/256, E, 8); streams 268 MB of W2 once, no atomics (deterministic).
__global__ __launch_bounds__(256)
void k_expert_mlp2(const float* __restrict__ mbuf, const float* __restrict__ W2,
                   float* __restrict__ part)
{
    const int dcol = blockIdx.x * 256 + threadIdx.x;
    const int e = blockIdx.y, fc = blockIdx.z;
    const float* w2 = W2 + (size_t)e * FFN * DMODEL + (size_t)fc * 512 * DMODEL + dcol;
    const float* m0 = mbuf + (size_t)(0 * 16 + e) * FFN + fc * 512;
    const float* m1 = mbuf + (size_t)(1 * 16 + e) * FFN + fc * 512;
    const float* m2 = mbuf + (size_t)(2 * 16 + e) * FFN + fc * 512;
    const float* m3 = mbuf + (size_t)(3 * 16 + e) * FFN + fc * 512;
    float a0 = 0, a1 = 0, a2 = 0, a3 = 0;
    #pragma unroll 4
    for (int f = 0; f < 512; ++f) {
        const float w = w2[(size_t)f * DMODEL];
        a0 += m0[f] * w; a1 += m1[f] * w; a2 += m2[f] * w; a3 += m3[f] * w;
    }
    const size_t base = (size_t)fc * 64 * DMODEL + dcol;
    part[base + (size_t)(0 * 16 + e) * DMODEL] = a0;
    part[base + (size_t)(1 * 16 + e) * DMODEL] = a1;
    part[base + (size_t)(2 * 16 + e) * DMODEL] = a2;
    part[base + (size_t)(3 * 16 + e) * DMODEL] = a3;
}

// o2 = o + b2[e] + sum_fc part
__global__ __launch_bounds__(256)
void k_expert_reduce(const float* __restrict__ obuf, const float* __restrict__ b2,
                     const float* __restrict__ part, float* __restrict__ o2)
{
    const int idx = blockIdx.x * 256 + threadIdx.x;   // 65536 total
    const int dd = idx & 1023, row = idx >> 10, e = row & 15;
    float acc = obuf[idx] + b2[e * DMODEL + dd];
    #pragma unroll
    for (int fc = 0; fc < 8; ++fc) acc += part[((size_t)fc * 64 + row) * DMODEL + dd];
    o2[idx] = acc;
}

// ------------------------------ combine attention: tokens attend to 16 slots
// one thread per (token, head); Kc/Vc (256 KB) live in L2. Output bf16.
__global__ __launch_bounds__(256)
void k_combine_attn(const float* __restrict__ Qc, const float* __restrict__ Kc,
                    const float* __restrict__ Vc, bf16* __restrict__ attnc)
{
    const int gid = blockIdx.x * 256 + threadIdx.x;   // TOK*16 threads
    const int h = gid & 15, token = gid >> 4, b = token >> 11;
    const float4* qr = (const float4*)(Qc + (size_t)token * DMODEL + h * 64);

    float p[16]; float mx = -3.0e38f;
    for (int s = 0; s < 16; ++s) {
        const float4* kr = (const float4*)(Kc + (size_t)(b * 16 + s) * DMODEL + h * 64);
        float a = 0.f;
        #pragma unroll
        for (int i = 0; i < 16; ++i) {
            const float4 qv = qr[i], kv = kr[i];
            a += qv.x * kv.x + qv.y * kv.y + qv.z * kv.z + qv.w * kv.w;
        }
        a *= 0.125f; p[s] = a; mx = fmaxf(mx, a);
    }
    float sum = 0.f;
    #pragma unroll
    for (int s = 0; s < 16; ++s) { p[s] = __expf(p[s] - mx); sum += p[s]; }
    const float inv = 1.f / sum;
    #pragma unroll
    for (int s = 0; s < 16; ++s) p[s] *= inv;

    bf16* dst = attnc + (size_t)token * DMODEL + h * 64;
    for (int i = 0; i < 16; ++i) {
        float ox = 0, oy = 0, oz = 0, ow = 0;
        #pragma unroll
        for (int s = 0; s < 16; ++s) {
            const float4 vv = ((const float4*)(Vc + (size_t)(b * 16 + s) * DMODEL + h * 64))[i];
            ox += p[s] * vv.x; oy += p[s] * vv.y; oz += p[s] * vv.z; ow += p[s] * vv.w;
        }
        v4bf o; o[0] = (bf16)ox; o[1] = (bf16)oy; o[2] = (bf16)oz; o[3] = (bf16)ow;
        *(v4bf*)(dst + i * 4) = o;
    }
}

// ============================================================== launcher
extern "C" void kernel_launch(void* const* d_in, const int* in_sizes, int n_in,
                              void* d_out, int out_size, void* d_ws, size_t ws_size,
                              hipStream_t stream)
{
    const float* x     = (const float*)d_in[0];
    const float* slotq = (const float*)d_in[1];
    const float* d_Wq  = (const float*)d_in[2];
    const float* d_Wk  = (const float*)d_in[3];
    const float* d_Wv  = (const float*)d_in[4];
    const float* d_Wo  = (const float*)d_in[5];
    const float* d_bq  = (const float*)d_in[6];
    const float* d_bk  = (const float*)d_in[7];
    const float* d_bv  = (const float*)d_in[8];
    const float* d_bo  = (const float*)d_in[9];
    const float* c_Wq  = (const float*)d_in[10];
    const float* c_Wk  = (const float*)d_in[11];
    const float* c_Wv  = (const float*)d_in[12];
    const float* c_Wo  = (const float*)d_in[13];
    const float* c_bq  = (const float*)d_in[14];
    const float* c_bk  = (const float*)d_in[15];
    const float* c_bv  = (const float*)d_in[16];
    const float* c_bo  = (const float*)d_in[17];
    const float* e_ln_g = (const float*)d_in[18];
    const float* e_ln_b = (const float*)d_in[19];
    const float* e_W1  = (const float*)d_in[20];
    const float* e_b1  = (const float*)d_in[21];
    const float* e_Wg  = (const float*)d_in[22];
    const float* e_bg  = (const float*)d_in[23];
    const float* e_W2  = (const float*)d_in[24];
    const float* e_b2  = (const float*)d_in[25];
    float* out = (float*)d_out;

    char* ws = (char*)d_ws;
    size_t off = 0;
    auto alloc = [&](size_t bytes) -> void* {
        void* p = ws + off;
        off += (bytes + 255) & ~(size_t)255;
        return p;
    };

    bf16*  xb    = (bf16*)alloc((size_t)TOK * DMODEL * 2);
    bf16*  wtKd  = (bf16*)alloc((size_t)DMODEL * DMODEL * 2);
    bf16*  wtVd  = (bf16*)alloc((size_t)DMODEL * DMODEL * 2);
    bf16*  wtQc  = (bf16*)alloc((size_t)DMODEL * DMODEL * 2);
    bf16*  wtOc  = (bf16*)alloc((size_t)DMODEL * DMODEL * 2);
    float* Kbuf  = (float*)alloc((size_t)TOK * DMODEL * 4);
    float* Vbuf  = (float*)alloc((size_t)TOK * DMODEL * 4);
    float* Qcbuf = (float*)alloc((size_t)TOK * DMODEL * 4);
    bf16*  attnc = (bf16*)alloc((size_t)TOK * DMODEL * 2);
    float* qh16  = (float*)alloc((size_t)16 * DMODEL * 4);
    float* attnd = (float*)alloc((size_t)64 * DMODEL * 4);
    float* obuf  = (float*)alloc((size_t)64 * DMODEL * 4);
    float* hn    = (float*)alloc((size_t)64 * DMODEL * 4);
    float* mbuf  = (float*)alloc((size_t)64 * FFN * 4);
    float* part  = (float*)alloc((size_t)8 * 64 * DMODEL * 4);
    float* o2    = (float*)alloc((size_t)64 * DMODEL * 4);
    float* Kc    = (float*)alloc((size_t)64 * DMODEL * 4);
    float* Vcb   = (float*)alloc((size_t)64 * DMODEL * 4);
    (void)ws_size; (void)in_sizes; (void)n_in; (void)out_size;

    const dim3 blk(256);
    const dim3 tgrid(32, 32);                       // 1024x1024 transpose
    const dim3 ggrid(DMODEL / BN, TOK / BM);        // (8, 64) WMMA blocks

    // 1. convert x to bf16
    k_cvt_bf16<<<(TOK * DMODEL / 4 + 255) / 256, blk, 0, stream>>>(x, xb, TOK * DMODEL / 4);
    // 2. transpose+convert the four big weights
    k_transpose_bf16<<<tgrid, blk, 0, stream>>>(d_Wk, wtKd, DMODEL, DMODEL);
    k_transpose_bf16<<<tgrid, blk, 0, stream>>>(d_Wv, wtVd, DMODEL, DMODEL);
    k_transpose_bf16<<<tgrid, blk, 0, stream>>>(c_Wq, wtQc, DMODEL, DMODEL);
    k_transpose_bf16<<<tgrid, blk, 0, stream>>>(c_Wo, wtOc, DMODEL, DMODEL);
    // 3. big WMMA projections (async-DMA double-buffered)
    k_gemm_bf16<<<ggrid, blk, 0, stream>>>(xb, wtKd, d_bk, Kbuf,  TOK, DMODEL, DMODEL);
    k_gemm_bf16<<<ggrid, blk, 0, stream>>>(xb, wtVd, d_bv, Vbuf,  TOK, DMODEL, DMODEL);
    k_gemm_bf16<<<ggrid, blk, 0, stream>>>(xb, wtQc, c_bq, Qcbuf, TOK, DMODEL, DMODEL);
    // 4. slot queries projection (16 rows, fp32)
    k_small_gemm<<<dim3(4, 16), blk, 0, stream>>>(slotq, d_Wq, d_bq, nullptr, 1, qh16, DMODEL, DMODEL);
    // 5. dispatch attention (flash softmax over 2048 tokens)
    k_dispatch_attn<<<BATCH * 16, blk, 0, stream>>>(qh16, Kbuf, Vbuf, attnd);
    // 6. o = attn @ Wo + bo + slot_query residual
    k_small_gemm<<<dim3(4, 64), blk, 0, stream>>>(attnd, d_Wo, d_bo, slotq, 16, obuf, DMODEL, DMODEL);
    // 7. expert input LayerNorm
    k_expert_ln<<<64, blk, 0, stream>>>(obuf, e_ln_g, e_ln_b, hn);
    // 8-10. gated expert MLPs (bandwidth-bound on 805 MB of weights)
    k_expert_mlp1<<<dim3(FFN / 256, 16), blk, 0, stream>>>(hn, e_W1, e_b1, e_Wg, e_bg, mbuf);
    k_expert_mlp2<<<dim3(DMODEL / 256, 16, 8), blk, 0, stream>>>(mbuf, e_W2, part);
    k_expert_reduce<<<(64 * DMODEL + 255) / 256, blk, 0, stream>>>(obuf, e_b2, part, o2);
    // 11. combine K/V projections of the 64 slot rows
    k_small_gemm<<<dim3(4, 64), blk, 0, stream>>>(o2, c_Wk, c_bk, nullptr, 1, Kc,  DMODEL, DMODEL);
    k_small_gemm<<<dim3(4, 64), blk, 0, stream>>>(o2, c_Wv, c_bv, nullptr, 1, Vcb, DMODEL, DMODEL);
    // 12. combine attention -> bf16 activations
    k_combine_attn<<<(TOK * 16 + 255) / 256, blk, 0, stream>>>(Qcbuf, Kc, Vcb, attnc);
    // 13. final WMMA output projection straight into d_out
    k_gemm_bf16<<<ggrid, blk, 0, stream>>>(attnc, wtOc, c_bo, out, TOK, DMODEL, DMODEL);
}
